// ALiBiTransformerLayer_43654047596564
// MI455X (gfx1250) — matrix operand order repaired and checked by reading, not compiled, not run on hardware
//
#include <hip/hip_runtime.h>

typedef __attribute__((ext_vector_type(16))) _Float16 v16h;
typedef __attribute__((ext_vector_type(8)))  float    v8f;

constexpr int kB  = 2;
constexpr int kS  = 2048;
constexpr int kD  = 1024;
constexpr int kH  = 16;
constexpr int kF  = 4096;
constexpr int kHD = 64;
constexpr int kBS = kB * kS;   // 4096 token rows

// ---------------------------------------------------------------------------
// CDNA5 async global->LDS copy (ASYNCcnt-tracked DMA path); fallback to
// regular VGPR-staged copies when the builtins are absent.
// Probe result: the builtin takes generic pointers to GCC int __vector(4).
// ---------------------------------------------------------------------------
#if defined(__has_builtin)
#  if __has_builtin(__builtin_amdgcn_global_load_async_to_lds_b128) && \
      __has_builtin(__builtin_amdgcn_s_wait_asynccnt)
#    define ATHENA_ASYNC 1
#  endif
#endif
#ifndef ATHENA_ASYNC
#  define ATHENA_ASYNC 0
#endif

typedef int av4i __attribute__((vector_size(16)));

__device__ __forceinline__ void async_cp16(const _Float16* __restrict__ g, _Float16* l) {
#if ATHENA_ASYNC
  __builtin_amdgcn_global_load_async_to_lds_b128((av4i*)g, (av4i*)l, 0, 0);
#else
  *(uint4*)l = *(const uint4*)g;
#endif
}

__device__ __forceinline__ void async_wait_all() {
#if ATHENA_ASYNC
  __builtin_amdgcn_s_wait_asynccnt(0);
#endif
}

// ---------------------------------------------------------------------------
// WMMA fragment helpers (wave32, 16x16x32 f16 -> f32)
// A 16x32 f16 layout (ISA 7.12.2): lane L (row = L&15):
//   half = L>>4 ; VGPR0..3 hold K = half*8 + {0..7}, VGPR4..7 hold K = 16+half*8+{0..7}
// B 32x16 f16 layout: lane L (col = L&15): K order as A (half*8+{0..7}, then 16+half*8+{0..7})
// C/D 16x16 f32: lane L holds col N=L&15, VGPR r holds row M = r + 8*(L>>4)
// ---------------------------------------------------------------------------
union HVec { v16h v; uint4 q[2]; };

__device__ __forceinline__ v16h ld_afrag(const _Float16* base, int ldh, int k0, int lane) {
  HVec u;
  const _Float16* rp = base + (lane & 15) * ldh + k0 + (lane >> 4) * 8;
  u.q[0] = *(const uint4*)(rp);        // K = kb .. kb+7
  u.q[1] = *(const uint4*)(rp + 16);   // K = kb+16 .. kb+23
  return u.v;
}

__device__ __forceinline__ v16h ld_bfrag(const _Float16* base, int ldh, int k0, int lane) {
  HVec u;
  const _Float16* rp = base + (lane & 15) * ldh + k0 + (lane >> 4) * 8;
  u.q[0] = *(const uint4*)(rp);        // K = kb .. kb+7
  u.q[1] = *(const uint4*)(rp + 16);   // K = kb+16 .. kb+23
  return u.v;
}

__device__ __forceinline__ v8f wmma32(v16h a, v16h b, v8f c) {
  return __builtin_amdgcn_wmma_f32_16x16x32_f16(false, a, false, b, (short)0, c,
                                                false, false);
}

// ---------------------------------------------------------------------------
// Elementwise fp32 -> fp16 (weights)
// ---------------------------------------------------------------------------
__global__ void k_f32_to_f16(const float* __restrict__ in, _Float16* __restrict__ out, int n) {
  int i = blockIdx.x * 256 + threadIdx.x;
  if (i < n) out[i] = (_Float16)in[i];
}

// ---------------------------------------------------------------------------
// LayerNorm (one row of D=1024 per block) -> f16
// ---------------------------------------------------------------------------
__global__ __launch_bounds__(256) void k_layernorm(const float* __restrict__ x,
                                                   const float* __restrict__ g,
                                                   const float* __restrict__ be,
                                                   _Float16* __restrict__ out) {
  const int row = blockIdx.x;
  const float* xr = x + (size_t)row * kD;
  const int tid = threadIdx.x, lane = tid & 31, w = tid >> 5;
  float s = 0.f, s2 = 0.f;
  #pragma unroll
  for (int i = tid; i < kD; i += 256) { float v = xr[i]; s += v; s2 += v * v; }
  #pragma unroll
  for (int m = 16; m >= 1; m >>= 1) {
    s  += __shfl_xor(s,  m, 32);
    s2 += __shfl_xor(s2, m, 32);
  }
  __shared__ float red[16];
  if (lane == 0) { red[w] = s; red[8 + w] = s2; }
  __syncthreads();
  if (tid == 0) {
    float a = 0.f, b2 = 0.f;
    for (int j = 0; j < 8; ++j) { a += red[j]; b2 += red[8 + j]; }
    red[0] = a; red[8] = b2;
  }
  __syncthreads();
  const float mean = red[0] * (1.0f / kD);
  const float var  = red[8] * (1.0f / kD) - mean * mean;
  const float inv  = rsqrtf(var + 1e-5f);
  _Float16* orow = out + (size_t)row * kD;
  for (int i = tid; i < kD; i += 256)
    orow[i] = (_Float16)((xr[i] - mean) * inv * g[i] + be[i]);
}

// ---------------------------------------------------------------------------
// v [B,S,D] (D = h*64+d) -> vt [B,H,HD,S], LDS-tiled so both global sides are
// coalesced. One block = one (b,h) x 64-row s-tile (64x64 elements).
// ---------------------------------------------------------------------------
__global__ __launch_bounds__(256) void k_transpose_v(const _Float16* __restrict__ v,
                                                     _Float16* __restrict__ vt) {
  __shared__ _Float16 tile[64][65];
  const int blk = blockIdx.x;            // B*H*(S/64) = 1024
  const int st  = blk & 31;              // s tile
  const int h   = (blk >> 5) & 15;
  const int b   = blk >> 9;
  const int tid = threadIdx.x;
  const int s0  = st * 64;
  // read: rows = s, cols = d (contiguous in global)
  #pragma unroll
  for (int it = 0; it < 16; ++it) {
    int idx = tid + it * 256;            // 0..4095
    int sr = idx >> 6, dc = idx & 63;
    tile[sr][dc] = v[(size_t)(b * kS + s0 + sr) * kD + h * kHD + dc];
  }
  __syncthreads();
  // write: rows = d, cols = s (contiguous in global)
  #pragma unroll
  for (int it = 0; it < 16; ++it) {
    int idx = tid + it * 256;
    int dr = idx >> 6, sc = idx & 63;
    vt[(size_t)((b * kH + h) * kHD + dr) * kS + s0 + sc] = tile[sc][dr];
  }
}

// ---------------------------------------------------------------------------
// WMMA GEMM: C[M,N] = A[M,K] * W[N,K]^T (+epilogue). A,W f16 row-major.
// Block tile 128x128, K-step 32, 256 threads = 8 waves; each wave: 32x64.
// Double-buffered LDS with async global->LDS copies overlapping compute.
// EPI 0: store f16. EPI 1: +bias, ReLU, store f16. EPI 2: +bias+resid, store f32.
// ---------------------------------------------------------------------------
template <int EPI>
__global__ __launch_bounds__(256) void k_gemm(const _Float16* __restrict__ A,
                                              const _Float16* __restrict__ W,
                                              const float* __restrict__ bias,
                                              const float* __restrict__ resid,
                                              _Float16* __restrict__ out16,
                                              float* __restrict__ out32,
                                              int M, int N, int K) {
  __shared__ _Float16 As[2][128 * 32];
  __shared__ _Float16 Ws[2][128 * 32];
  const int tid = threadIdx.x, lane = tid & 31, w = tid >> 5;
  const int wm = w & 3, wn = w >> 2;
  const int bm = blockIdx.y * 128, bn = blockIdx.x * 128;
  (void)M;

  auto issue_tile = [&](int buf, int k0) {
    #pragma unroll
    for (int it = 0; it < 2; ++it) {
      int idx = (tid + it * 256) * 8;
      int row = idx >> 5, col = idx & 31;
      async_cp16(A + (size_t)(bm + row) * K + k0 + col, &As[buf][row * 32 + col]);
      async_cp16(W + (size_t)(bn + row) * K + k0 + col, &Ws[buf][row * 32 + col]);
    }
  };

  v8f acc[2][4];
  const v8f vz = {0.f, 0.f, 0.f, 0.f, 0.f, 0.f, 0.f, 0.f};
  #pragma unroll
  for (int mt = 0; mt < 2; ++mt)
    #pragma unroll
    for (int nt = 0; nt < 4; ++nt) acc[mt][nt] = vz;

  issue_tile(0, 0);
  async_wait_all();
  __syncthreads();

  int buf = 0;
  for (int k0 = 0; k0 < K; k0 += 32, buf ^= 1) {
    if (k0 + 32 < K) {
      issue_tile(buf ^ 1, k0 + 32);     // DMA next tile while we compute
    }
#if !ATHENA_ASYNC
    if (k0 + 32 < K) {                  // fallback: at least prefetch to cache
      #pragma unroll
      for (int it = 0; it < 2; ++it) {
        int idx = (tid + it * 256) * 8;
        int row = idx >> 5, col = idx & 31;
        __builtin_prefetch(A + (size_t)(bm + row) * K + k0 + 32 + col, 0, 1);
      }
    }
#endif
    v16h af[2], bf[4];
    #pragma unroll
    for (int mt = 0; mt < 2; ++mt)
      af[mt] = ld_afrag(&As[buf][(wm * 32 + mt * 16) * 32], 32, 0, lane);
    #pragma unroll
    for (int nt = 0; nt < 4; ++nt)
      bf[nt] = ld_bfrag(&Ws[buf][(wn * 64 + nt * 16) * 32], 32, 0, lane);
    #pragma unroll
    for (int mt = 0; mt < 2; ++mt)
      #pragma unroll
      for (int nt = 0; nt < 4; ++nt)
        acc[mt][nt] = wmma32(af[mt], bf[nt], acc[mt][nt]);

    async_wait_all();                   // this wave's copies into buf^1 done
    __syncthreads();                    // everyone done reading buf
  }

  const int coll = lane & 15, rofs = (lane >> 4) * 8;
  #pragma unroll
  for (int mt = 0; mt < 2; ++mt) {
    #pragma unroll
    for (int r = 0; r < 8; ++r) {
      const int row = bm + wm * 32 + mt * 16 + r + rofs;
      #pragma unroll
      for (int nt = 0; nt < 4; ++nt) {
        const int col = bn + wn * 64 + nt * 16 + coll;
        float v = acc[mt][nt][r];
        if (EPI == 1) { v += bias[col]; v = fmaxf(v, 0.0f); }
        if (EPI == 2) { v += bias[col] + resid[(size_t)row * N + col]; }
        if (EPI == 2) out32[(size_t)row * N + col] = v;
        else          out16[(size_t)row * N + col] = (_Float16)v;
      }
    }
  }
}

// ---------------------------------------------------------------------------
// Fused ALiBi attention (flash style, non-causal).
// Grid: B*H*(S/64) blocks, 128 threads (4 waves); wave w owns 16 query rows.
// Streams 128-key blocks: S = Q·K^T (16 WMMAs/wave), online softmax with
// ALiBi bias slope*(j-i), P re-laid-out via private LDS, O += P·V (16 WMMAs).
// K/V tiles arrive via async global->LDS DMA.
// ---------------------------------------------------------------------------
__global__ __launch_bounds__(128) void k_flash_alibi(const _Float16* __restrict__ q,
                                                     const _Float16* __restrict__ k,
                                                     const _Float16* __restrict__ vt,
                                                     _Float16* __restrict__ ao) {
  __shared__ _Float16 Qs[64 * 64];
  __shared__ _Float16 Ks[128 * 64];
  __shared__ _Float16 Vts[64 * 128];   // V^T tile: [d][j]
  __shared__ _Float16 Ps[4][16 * 128]; // per-wave P panel

  const int blk  = blockIdx.x;
  const int qblk = blk & 31;           // S/64 = 32
  const int h    = (blk >> 5) & 15;
  const int b    = blk >> 9;
  const int tid = threadIdx.x, lane = tid & 31, w = tid >> 5;
  const int q0 = qblk * 64;
  const int coll = lane & 15;
  const float slope = exp2f(-0.5f * (float)(h + 1));  // 2^(-(h+1)/2) for H=16

  #pragma unroll
  for (int it = 0; it < 4; ++it) {     // Q block 64x64
    int idx = (tid + it * 128) * 8;
    int row = idx >> 6, col = idx & 63;
    async_cp16(q + (size_t)(b * kS + q0 + row) * kD + h * kHD + col,
               &Qs[row * 64 + col]);
  }
  async_wait_all();
  __syncthreads();
  v16h qf[2];
  qf[0] = ld_afrag(&Qs[(w * 16) * 64], 64, 0, lane);
  qf[1] = ld_afrag(&Qs[(w * 16) * 64], 64, 32, lane);

  const v8f vz = {0.f, 0.f, 0.f, 0.f, 0.f, 0.f, 0.f, 0.f};
  float m[8], l[8], fi[8];
  v8f o[4];
  #pragma unroll
  for (int r = 0; r < 8; ++r) {
    m[r] = -1e30f; l[r] = 0.f;
    fi[r] = (float)(q0 + w * 16 + r + (lane >> 4) * 8);
  }
  #pragma unroll
  for (int t = 0; t < 4; ++t) o[t] = vz;
  _Float16* pw = &Ps[w][0];

  for (int jb = 0; jb < kS / 128; ++jb) {
    const int j0 = jb * 128;
    __syncthreads();                   // everyone done reading Ks/Vts
    #pragma unroll
    for (int it = 0; it < 8; ++it) {   // K block 128x64
      int idx = (tid + it * 128) * 8;
      int row = idx >> 6, col = idx & 63;
      async_cp16(k + (size_t)(b * kS + j0 + row) * kD + h * kHD + col,
                 &Ks[row * 64 + col]);
    }
    #pragma unroll
    for (int it = 0; it < 8; ++it) {   // V^T block 64x128 (contiguous along j)
      int idx = (tid + it * 128) * 8;
      int d = idx >> 7, j = idx & 127;
      async_cp16(vt + (size_t)((b * kH + h) * kHD + d) * kS + j0 + j,
                 &Vts[d * 128 + j]);
    }
    async_wait_all();
    __syncthreads();

    // ---- scores: 8 tiles of 16x16, K=64 split in two 32-steps ----
    v8f st[8];
    #pragma unroll
    for (int jt = 0; jt < 8; ++jt) {
      v16h b0 = ld_bfrag(&Ks[(jt * 16) * 64], 64, 0, lane);
      v8f t = wmma32(qf[0], b0, vz);
      v16h b1 = ld_bfrag(&Ks[(jt * 16) * 64], 64, 32, lane);
      st[jt] = wmma32(qf[1], b1, t);
    }

    // ---- ALiBi bias + running max ----
    float nm[8];
    #pragma unroll
    for (int r = 0; r < 8; ++r) nm[r] = m[r];
    #pragma unroll
    for (int jt = 0; jt < 8; ++jt) {
      const float jf = (float)(j0 + jt * 16 + coll);
      #pragma unroll
      for (int r = 0; r < 8; ++r) {
        float v = st[jt][r] * 0.125f + slope * (jf - fi[r]);
        st[jt][r] = v;
        nm[r] = fmaxf(nm[r], v);
      }
    }
    #pragma unroll
    for (int r = 0; r < 8; ++r) {      // row max across lane-half (16 lanes)
      nm[r] = fmaxf(nm[r], __shfl_xor(nm[r], 1, 32));
      nm[r] = fmaxf(nm[r], __shfl_xor(nm[r], 2, 32));
      nm[r] = fmaxf(nm[r], __shfl_xor(nm[r], 4, 32));
      nm[r] = fmaxf(nm[r], __shfl_xor(nm[r], 8, 32));
    }
    float rs[8];
    #pragma unroll
    for (int r = 0; r < 8; ++r) {      // rescale running state
      float sc = __expf(m[r] - nm[r]);
      m[r] = nm[r];
      l[r] *= sc;
      rs[r] = 0.f;
      #pragma unroll
      for (int t = 0; t < 4; ++t) o[t][r] = o[t][r] * sc;
    }
    #pragma unroll
    for (int jt = 0; jt < 8; ++jt) {   // exp, rowsum, write P panel (C->A relayout)
      const int col = jt * 16 + coll;
      #pragma unroll
      for (int r = 0; r < 8; ++r) {
        float pv = __expf(st[jt][r] - m[r]);
        rs[r] += pv;
        int row = r + (lane >> 4) * 8;
        pw[row * 128 + col] = (_Float16)pv;
      }
    }
    #pragma unroll
    for (int r = 0; r < 8; ++r) {
      rs[r] += __shfl_xor(rs[r], 1, 32);
      rs[r] += __shfl_xor(rs[r], 2, 32);
      rs[r] += __shfl_xor(rs[r], 4, 32);
      rs[r] += __shfl_xor(rs[r], 8, 32);
      l[r] += rs[r];
    }

    // ---- O += P (16x128) * V (128x64) ----
    #pragma unroll
    for (int ks = 0; ks < 4; ++ks) {
      v16h pa = ld_afrag(pw, 128, ks * 32, lane);
      #pragma unroll
      for (int dt = 0; dt < 4; ++dt) {
        v16h vb = ld_bfrag(&Vts[(dt * 16) * 128], 128, ks * 32, lane);
        o[dt] = wmma32(pa, vb, o[dt]);
      }
    }
  }

  #pragma unroll
  for (int r = 0; r < 8; ++r) {        // normalize + store ao [B,S,D]
    const float inv = 1.0f / l[r];
    const int row = q0 + w * 16 + r + (lane >> 4) * 8;
    const size_t base = (size_t)(b * kS + row) * kD + h * kHD;
    #pragma unroll
    for (int dt = 0; dt < 4; ++dt)
      ao[base + dt * 16 + coll] = (_Float16)(o[dt][r] * inv);
  }
}

// ---------------------------------------------------------------------------
// Host-side orchestration
// ---------------------------------------------------------------------------
extern "C" void kernel_launch(void* const* d_in, const int* in_sizes, int n_in,
                              void* d_out, int out_size, void* d_ws, size_t ws_size,
                              hipStream_t stream) {
  (void)in_sizes; (void)n_in; (void)out_size; (void)ws_size;
  const float* x   = (const float*)d_in[0];
  const float* Wq  = (const float*)d_in[1];
  const float* Wk  = (const float*)d_in[2];
  const float* Wv  = (const float*)d_in[3];
  const float* Wo  = (const float*)d_in[4];
  const float* bo  = (const float*)d_in[5];
  const float* W1  = (const float*)d_in[6];
  const float* b1  = (const float*)d_in[7];
  const float* W2  = (const float*)d_in[8];
  const float* b2  = (const float*)d_in[9];
  const float* g1  = (const float*)d_in[10];
  const float* be1 = (const float*)d_in[11];
  const float* g2  = (const float*)d_in[12];
  const float* be2 = (const float*)d_in[13];
  float* out = (float*)d_out;

  char* p = (char*)d_ws;
  auto alloc = [&](size_t bytes) -> char* {
    char* r = p;
    p += (bytes + 255) & ~(size_t)255;
    return r;
  };
  _Float16* h16   = (_Float16*)alloc((size_t)kBS * kD * 2);
  _Float16* wq16  = (_Float16*)alloc((size_t)kD * kD * 2);
  _Float16* wk16  = (_Float16*)alloc((size_t)kD * kD * 2);
  _Float16* wv16  = (_Float16*)alloc((size_t)kD * kD * 2);
  _Float16* wo16  = (_Float16*)alloc((size_t)kD * kD * 2);
  _Float16* w116  = (_Float16*)alloc((size_t)kF * kD * 2);
  _Float16* w216  = (_Float16*)alloc((size_t)kD * kF * 2);
  _Float16* q16   = (_Float16*)alloc((size_t)kBS * kD * 2);
  _Float16* k16   = (_Float16*)alloc((size_t)kBS * kD * 2);
  _Float16* v16   = (_Float16*)alloc((size_t)kBS * kD * 2);
  _Float16* vt16  = (_Float16*)alloc((size_t)kB * kH * kHD * kS * 2);
  _Float16* ao16  = (_Float16*)alloc((size_t)kBS * kD * 2);
  float*    x2    = (float*)   alloc((size_t)kBS * kD * 4);
  _Float16* h216  = (_Float16*)alloc((size_t)kBS * kD * 2);
  _Float16* ff16  = (_Float16*)alloc((size_t)kBS * kF * 2);

  // weight conversions
  const int nw = kD * kD, nf = kF * kD;
  k_f32_to_f16<<<(nw + 255) / 256, 256, 0, stream>>>(Wq, wq16, nw);
  k_f32_to_f16<<<(nw + 255) / 256, 256, 0, stream>>>(Wk, wk16, nw);
  k_f32_to_f16<<<(nw + 255) / 256, 256, 0, stream>>>(Wv, wv16, nw);
  k_f32_to_f16<<<(nw + 255) / 256, 256, 0, stream>>>(Wo, wo16, nw);
  k_f32_to_f16<<<(nf + 255) / 256, 256, 0, stream>>>(W1, w116, nf);
  k_f32_to_f16<<<(nf + 255) / 256, 256, 0, stream>>>(W2, w216, nf);

  // pre-norm 1 -> h16
  k_layernorm<<<kBS, 256, 0, stream>>>(x, g1, be1, h16);

  // QKV projections
  dim3 gproj(kD / 128, kBS / 128);
  k_gemm<0><<<gproj, 256, 0, stream>>>(h16, wq16, nullptr, nullptr, q16, nullptr, kBS, kD, kD);
  k_gemm<0><<<gproj, 256, 0, stream>>>(h16, wk16, nullptr, nullptr, k16, nullptr, kBS, kD, kD);
  k_gemm<0><<<gproj, 256, 0, stream>>>(h16, wv16, nullptr, nullptr, v16, nullptr, kBS, kD, kD);

  // V transpose for contiguous P·V B-fragments
  k_transpose_v<<<kB * kH * (kS / 64), 256, 0, stream>>>(v16, vt16);

  // fused ALiBi attention
  k_flash_alibi<<<kB * kH * (kS / 64), 128, 0, stream>>>(q16, k16, vt16, ao16);

  // out-proj + residual: x2 = x + ao·Wo^T + bo
  k_gemm<2><<<gproj, 256, 0, stream>>>(ao16, wo16, bo, x, nullptr, x2, kBS, kD, kD);

  // pre-norm 2 -> h216
  k_layernorm<<<kBS, 256, 0, stream>>>(x2, g2, be2, h216);

  // FFN
  dim3 gff1(kF / 128, kBS / 128);
  k_gemm<1><<<gff1, 256, 0, stream>>>(h216, w116, b1, nullptr, ff16, nullptr, kBS, kF, kD);
  k_gemm<2><<<gproj, 256, 0, stream>>>(ff16, w216, b2, x2, nullptr, out, kBS, kD, kF);
}
